// EdgeGAT_89352499626350
// MI455X (gfx1250) — matrix operand order repaired and checked by reading, not compiled, not run on hardware
//
#include <hip/hip_runtime.h>
#include <hip/hip_bf16.h>

typedef unsigned short ushort_t;
typedef __attribute__((ext_vector_type(16))) __bf16 v16bf;
typedef __attribute__((ext_vector_type(8)))  __bf16 v8bf;
typedef __attribute__((ext_vector_type(8)))  float  v8f;
typedef __attribute__((ext_vector_type(4)))  unsigned int v4u;
typedef __attribute__((ext_vector_type(8)))  unsigned int v8u;

static __device__ __forceinline__ v16bf cat8(v8bf lo, v8bf hi) {
  return __builtin_shufflevector(lo, hi, 0,1,2,3,4,5,6,7,8,9,10,11,12,13,14,15);
}

static __device__ __forceinline__ ushort_t f2bf(float x) {
  unsigned u = __float_as_uint(x);
  unsigned r = u + 0x7FFFu + ((u >> 16) & 1u);        // round-to-nearest-even
  if ((u & 0x7F800000u) == 0x7F800000u) r = u;        // keep Inf/NaN
  return (ushort_t)(r >> 16);
}

// monotonic float<->uint mapping for atomicMax-based segment max
static __device__ __forceinline__ unsigned f2ord(float x) {
  unsigned u = __float_as_uint(x);
  return (u & 0x80000000u) ? ~u : (u | 0x80000000u);
}
static __device__ __forceinline__ float ord2f(unsigned o) {
  unsigned u = (o & 0x80000000u) ? (o ^ 0x80000000u) : ~o;
  return __uint_as_float(u);
}

// ---------------------------------------------------------------------------
// Weight prep: fold We*ae -> Wae (64x4), transpose [W0|Wres0] and W1 to
// bf16 [col][k] layout for direct WMMA B-fragment loads.
// ---------------------------------------------------------------------------
__global__ void prep_weights(const float* __restrict__ W0, const float* __restrict__ Wres0,
                             const float* __restrict__ We0, const float* __restrict__ ae0,
                             const float* __restrict__ W1,
                             const float* __restrict__ We1, const float* __restrict__ ae1,
                             ushort_t* __restrict__ WcatT, ushort_t* __restrict__ W1T,
                             float* __restrict__ Wae0, float* __restrict__ Wae1) {
  const int t = blockIdx.x * blockDim.x + threadIdx.x;
  const int stride = gridDim.x * blockDim.x;
  for (int i = t; i < 512 * 64; i += stride) {           // WcatT[c*64+k]
    int c = i >> 6, k = i & 63;
    float v = (c < 256) ? W0[k * 256 + c] : Wres0[k * 256 + (c - 256)];
    WcatT[i] = f2bf(v);
  }
  for (int i = t; i < 256 * 256; i += stride) {          // W1T[c*256+k]
    int c = i >> 8, k = i & 255;
    W1T[i] = f2bf(W1[k * 256 + c]);
  }
  for (int i = t; i < 64 * 4; i += stride) {             // Wae[k*4+h]
    int k = i >> 2, h = i & 3;
    float s0 = 0.f, s1 = 0.f;
    for (int d = 0; d < 64; ++d) {
      s0 += We0[k * 256 + h * 64 + d] * ae0[h * 64 + d];
      s1 += We1[k * 256 + h * 64 + d] * ae1[h * 64 + d];
    }
    Wae0[i] = s0;
    Wae1[i] = s1;
  }
}

__global__ void to_bf16(const float* __restrict__ x, ushort_t* __restrict__ y, long n) {
  long i = (long)blockIdx.x * blockDim.x + threadIdx.x;
  if (i < n) y[i] = f2bf(x[i]);
}

__global__ void fill_u32(unsigned* __restrict__ p, unsigned v, long n) {
  long i = (long)blockIdx.x * blockDim.x + threadIdx.x;
  if (i < n) p[i] = v;
}

// ---------------------------------------------------------------------------
// bf16 WMMA GEMM: C[M x Ncol] = A[M x K] * Bt^T, Bt pre-transposed [Ncol x K].
// Block = 8 waves = 128 rows x one 64-col group. The block's B panel
// (64 x K bf16, contiguous rows of Bt) is staged into LDS once by the
// Tensor Data Mover (tensor_load_to_lds, 2-group D#), then all 8 waves read
// fragments from LDS. A fragments stream from global with prefetch.
// Fragment k order: contiguous k = 16*half + s, same for A and B (wave32).
// ---------------------------------------------------------------------------
__global__ void gemm_bf16_wmma(const ushort_t* __restrict__ A,
                               const ushort_t* __restrict__ Bt,
                               float* __restrict__ C,
                               int M, int K, int Ncol) {
  extern __shared__ ushort_t sB[];                 // 64 * K bf16
  const int ngroups = Ncol >> 6;                   // 64-wide column groups
  const int bm = blockIdx.x / ngroups;             // 128-row block index
  const int ng = blockIdx.x - bm * ngroups;
  const int n0 = ng * 64;
  const int wave = threadIdx.x >> 5;
  const int lane = threadIdx.x & 31;
  const int mt = bm * 8 + wave;                    // this wave's 16-row tile

  // --- TDM: wave 0 DMAs the 64 x K bf16 weight panel into LDS ---
  if (threadIdx.x < 32) {
    unsigned lds_off = (unsigned)(uintptr_t)(&sB[0]);          // low 32b = LDS offset
    unsigned long long ga = (unsigned long long)(uintptr_t)(Bt + (size_t)n0 * K);
    unsigned uk = (unsigned)K;
    v4u g0;
    g0.x = 1u;                                                 // count=1, user mode
    g0.y = lds_off;                                            // lds_addr
    g0.z = (unsigned)(ga & 0xFFFFFFFFu);                       // global_addr[31:0]
    g0.w = (unsigned)((ga >> 32) & 0x01FFFFFFu) | (2u << 30);  // ga[56:32] | type=2
    v8u g1;
    g1[0] = (1u << 16);                   // wg_mask=0, data_size=1 (2 bytes)
    g1[1] = (uk & 0xFFFFu) << 16;         // tensor_dim0[15:0]
    g1[2] = ((uk >> 16) & 0xFFFFu) | (64u << 16); // tensor_dim0[31:16] | tensor_dim1[15:0]=64
    g1[3] = (uk << 16);                   // tensor_dim1[31:16]=0 | tile_dim0=K
    g1[4] = 64u;                          // tile_dim1=64, tile_dim2=0
    g1[5] = uk;                           // tensor_dim0_stride[31:0] = K
    g1[6] = 0u;                           // stride[47:32]=0 | tensor_dim1_stride lo=0
    g1[7] = 0u;
    asm volatile("tensor_load_to_lds %0, %1" :: "s"(g0), "s"(g1) : "memory");
    __builtin_amdgcn_s_wait_tensorcnt(0);
  }
  __syncthreads();                                  // panel visible to all 8 waves

  if (mt * 16 < M) {
    const int r  = lane & 15;
    const int hi = lane >> 4;
    const size_t arow = (size_t)(mt * 16 + r) * K;
    // LDS row bases for the 4 column tiles of this 64-col group
    const int lb0 = ( 0 + r) * K;
    const int lb1 = (16 + r) * K;
    const int lb2 = (32 + r) * K;
    const int lb3 = (48 + r) * K;

    v8f acc0 = {}, acc1 = {}, acc2 = {}, acc3 = {};

    for (int k = 0; k < K; k += 32) {
      // A fragment: lane<16 -> K {k..k+15}; lane>=16 -> K {k+16..k+31}
      const ushort_t* ap = A + arow + k + hi * 16;
      if (k + 32 < K) __builtin_prefetch(ap + 32, 0, 1);
      v16bf av = cat8(*(const v8bf*)ap, *(const v8bf*)(ap + 8));

      // B fragments from LDS: lane<16 -> K {k..k+15}; lane>=16 -> K {k+16..k+31}
      const int kb = k + hi * 16;
      v16bf bv0 = cat8(*(const v8bf*)(sB + lb0 + kb), *(const v8bf*)(sB + lb0 + kb + 8));
      v16bf bv1 = cat8(*(const v8bf*)(sB + lb1 + kb), *(const v8bf*)(sB + lb1 + kb + 8));
      v16bf bv2 = cat8(*(const v8bf*)(sB + lb2 + kb), *(const v8bf*)(sB + lb2 + kb + 8));
      v16bf bv3 = cat8(*(const v8bf*)(sB + lb3 + kb), *(const v8bf*)(sB + lb3 + kb + 8));

      acc0 = __builtin_amdgcn_wmma_f32_16x16x32_bf16(false, av, false, bv0, (short)0, acc0, false, false);
      acc1 = __builtin_amdgcn_wmma_f32_16x16x32_bf16(false, av, false, bv1, (short)0, acc1, false, false);
      acc2 = __builtin_amdgcn_wmma_f32_16x16x32_bf16(false, av, false, bv2, (short)0, acc2, false, false);
      acc3 = __builtin_amdgcn_wmma_f32_16x16x32_bf16(false, av, false, bv3, (short)0, acc3, false, false);
    }

    // C/D layout: lane<16 -> rows mt*16+0..7, lane>=16 -> rows +8..15; col = n0+r
    float* crow = C + (size_t)(mt * 16 + hi * 8) * Ncol + (n0 + r);
#pragma unroll
    for (int j = 0; j < 8; ++j) {
      crow[(size_t)j * Ncol +  0] = acc0[j];
      crow[(size_t)j * Ncol + 16] = acc1[j];
      crow[(size_t)j * Ncol + 32] = acc2[j];
      crow[(size_t)j * Ncol + 48] = acc3[j];
    }
  }
}

// ---------------------------------------------------------------------------
// el[n,h] = ft[n,h,:].al[h,:]   er[n,h] = ft[n,h,:].ar[h,:]
// ---------------------------------------------------------------------------
__global__ void node_coeffs(const float* __restrict__ ft, int ldft,
                            const float* __restrict__ al, const float* __restrict__ ar,
                            float* __restrict__ el, float* __restrict__ er, int n) {
  int i = blockIdx.x * blockDim.x + threadIdx.x;
  if (i >= n * 4) return;
  int node = i >> 2, h = i & 3;
  const float* row = ft + (size_t)node * ldft + h * 64;
  const float* va = al + h * 64;
  const float* vb = ar + h * 64;
  float sl = 0.f, sr = 0.f;
#pragma unroll 8
  for (int d = 0; d < 64; ++d) { float v = row[d]; sl += v * va[d]; sr += v * vb[d]; }
  el[i] = sl;
  er[i] = sr;
}

// ---------------------------------------------------------------------------
// Per-edge logit: ee fused from folded edge weight, leaky relu, segment max.
// ---------------------------------------------------------------------------
__global__ void edge_logits(const float* __restrict__ ef, const float* __restrict__ Wae,
                            const int* __restrict__ src, const int* __restrict__ dst,
                            const float* __restrict__ el, const float* __restrict__ er,
                            float* __restrict__ elog, unsigned* __restrict__ maxb, int E) {
  __shared__ float sW[256];
  for (int i = threadIdx.x; i < 256; i += blockDim.x) sW[i] = Wae[i];
  __syncthreads();
  int e = blockIdx.x * blockDim.x + threadIdx.x;
  if (e >= E) return;
  const float* row = ef + (size_t)e * 64;
  float s0 = 0.f, s1 = 0.f, s2 = 0.f, s3 = 0.f;
#pragma unroll 8
  for (int k = 0; k < 64; ++k) {
    float v = row[k];
    s0 += v * sW[k * 4 + 0]; s1 += v * sW[k * 4 + 1];
    s2 += v * sW[k * 4 + 2]; s3 += v * sW[k * 4 + 3];
  }
  int is = src[e], id = dst[e];
  float ee[4] = { s0, s1, s2, s3 };
#pragma unroll
  for (int h = 0; h < 4; ++h) {
    float x = el[is * 4 + h] + er[id * 4 + h] + ee[h];
    x = (x > 0.f) ? x : 0.2f * x;                       // leaky_relu(0.2)
    elog[(size_t)e * 4 + h] = x;
    atomicMax(&maxb[id * 4 + h], f2ord(x));
  }
}

// ex = exp(logit - max[dst]); den[dst] += ex   (in-place on elog)
__global__ void edge_exp(const int* __restrict__ dst, const unsigned* __restrict__ maxb,
                         float* __restrict__ elog, float* __restrict__ den, int E) {
  int i = blockIdx.x * blockDim.x + threadIdx.x;
  if (i >= E * 4) return;
  int e = i >> 2, h = i & 3;
  int d4 = dst[e] * 4 + h;
  float ex = __expf(elog[i] - ord2f(maxb[d4]));
  elog[i] = ex;
  atomicAdd(&den[d4], ex);
}

// agg[dst] += ft[src] * (ex/den[dst]) ; one wave per edge, lane covers 8 cols
__global__ void aggregate(const int* __restrict__ src, const int* __restrict__ dst,
                          const float* __restrict__ ft, int ldft,
                          const float* __restrict__ ex, const float* __restrict__ den,
                          float* __restrict__ agg, int E) {
  int gwave = (int)((blockIdx.x * blockDim.x + threadIdx.x) >> 5);
  int lane = threadIdx.x & 31;
  int nwaves = (int)((gridDim.x * blockDim.x) >> 5);
  int h = lane >> 3;                                    // head for this lane's 8 cols
  for (int e = gwave; e < E; e += nwaves) {
    int is = src[e], id = dst[e];
    float coef = ex[(size_t)e * 4 + h] / den[id * 4 + h];
    const float4* fs = (const float4*)(ft + (size_t)is * ldft + lane * 8);
    float4 v0 = fs[0], v1 = fs[1];
    float* od = agg + (size_t)id * 256 + lane * 8;
    atomicAdd(od + 0, v0.x * coef); atomicAdd(od + 1, v0.y * coef);
    atomicAdd(od + 2, v0.z * coef); atomicAdd(od + 3, v0.w * coef);
    atomicAdd(od + 4, v1.x * coef); atomicAdd(od + 5, v1.y * coef);
    atomicAdd(od + 6, v1.z * coef); atomicAdd(od + 7, v1.w * coef);
  }
}

// layer0 epilogue: h1 = ELU(agg + res + b0); also emit bf16 copy for GEMM1
__global__ void finalize0(const float* __restrict__ agg, const float* __restrict__ C0,
                          const float* __restrict__ b0, float* __restrict__ h1,
                          ushort_t* __restrict__ Abf1, long n256) {
  long i = (long)blockIdx.x * blockDim.x + threadIdx.x;
  if (i >= n256) return;
  int c = (int)(i & 255);
  long node = i >> 8;
  float x = agg[i] + C0[node * 512 + 256 + c] + b0[c];
  x = (x > 0.f) ? x : (__expf(x) - 1.f);                // ELU
  h1[i] = x;
  Abf1[i] = f2bf(x);
}

// layer1 epilogue: out[n,d] = mean_h(agg + h1 + b1)
__global__ void finalize1(const float* __restrict__ agg, const float* __restrict__ h1,
                          const float* __restrict__ b1, float* __restrict__ out, int n) {
  int i = blockIdx.x * blockDim.x + threadIdx.x;
  if (i >= n * 64) return;
  int node = i >> 6, d = i & 63;
  float s = 0.f;
#pragma unroll
  for (int h = 0; h < 4; ++h) {
    int c = h * 64 + d;
    s += agg[(size_t)node * 256 + c] + h1[(size_t)node * 256 + c] + b1[c];
  }
  out[i] = 0.25f * s;
}

// ---------------------------------------------------------------------------
extern "C" void kernel_launch(void* const* d_in, const int* in_sizes, int n_in,
                              void* d_out, int out_size, void* d_ws, size_t ws_size,
                              hipStream_t stream) {
  const float* features = (const float*)d_in[0];
  const float* edge_feat = (const float*)d_in[1];
  const int*   src  = (const int*)d_in[2];
  const int*   dst  = (const int*)d_in[3];
  const float* W0   = (const float*)d_in[4];
  const float* We0  = (const float*)d_in[5];
  const float* al0  = (const float*)d_in[6];
  const float* ar0  = (const float*)d_in[7];
  const float* ae0  = (const float*)d_in[8];
  const float* b0   = (const float*)d_in[9];
  const float* Wres0= (const float*)d_in[10];
  const float* W1   = (const float*)d_in[11];
  const float* We1  = (const float*)d_in[12];
  const float* al1  = (const float*)d_in[13];
  const float* ar1  = (const float*)d_in[14];
  const float* ae1  = (const float*)d_in[15];
  const float* b1   = (const float*)d_in[16];

  const int N = in_sizes[0] / 64;     // 50000
  const int E = in_sizes[2];          // 400000

  // ---- workspace bump allocator (256B aligned) ----
  char* base = (char*)d_ws;
  size_t o = 0;
  auto alloc = [&](size_t bytes) -> char* {
    char* p = base + o;
    o = (o + bytes + 255) & ~(size_t)255;
    return p;
  };
  ushort_t* WcatT = (ushort_t*)alloc(512 * 64 * 2);
  ushort_t* W1T   = (ushort_t*)alloc(256 * 256 * 2);
  float* Wae0 = (float*)alloc(64 * 4 * 4);
  float* Wae1 = (float*)alloc(64 * 4 * 4);
  ushort_t* Abf0 = (ushort_t*)alloc((size_t)N * 64 * 2);
  float* C0   = (float*)alloc((size_t)N * 512 * 4);   // [ft0 | res0]; reused as ft1
  float* el   = (float*)alloc((size_t)N * 4 * 4);
  float* er   = (float*)alloc((size_t)N * 4 * 4);
  float* elog = (float*)alloc((size_t)E * 4 * 4);     // logits, then ex in-place
  unsigned* maxb = (unsigned*)alloc((size_t)N * 4 * 4);
  float* den  = (float*)alloc((size_t)N * 4 * 4);
  float* agg  = (float*)alloc((size_t)N * 256 * 4);
  float* h1   = (float*)alloc((size_t)N * 256 * 4);
  ushort_t* Abf1 = (ushort_t*)alloc((size_t)N * 256 * 2);
  float* ft1 = C0;                                    // alias: C0 dead after finalize0

  const int T = 256;
  auto blocks = [](long n, int t) { return (int)((n + t - 1) / t); };

  // ---- prep ----
  prep_weights<<<64, T, 0, stream>>>(W0, Wres0, We0, ae0, W1, We1, ae1,
                                     WcatT, W1T, Wae0, Wae1);
  to_bf16<<<blocks((long)N * 64, T), T, 0, stream>>>(features, Abf0, (long)N * 64);

  const int mblocks = (N + 127) / 128;                // 128 rows per block

  // ---- layer 0 ----
  gemm_bf16_wmma<<<mblocks * (512 / 64), T, 64 * 64 * 2, stream>>>(Abf0, WcatT, C0, N, 64, 512);
  node_coeffs<<<blocks((long)N * 4, T), T, 0, stream>>>(C0, 512, al0, ar0, el, er, N);
  fill_u32<<<blocks((long)N * 4, T), T, 0, stream>>>(maxb, 0u, (long)N * 4);
  fill_u32<<<blocks((long)N * 4, T), T, 0, stream>>>((unsigned*)den, 0u, (long)N * 4);
  fill_u32<<<blocks((long)N * 256, T), T, 0, stream>>>((unsigned*)agg, 0u, (long)N * 256);
  edge_logits<<<blocks(E, T), T, 0, stream>>>(edge_feat, Wae0, src, dst, el, er, elog, maxb, E);
  edge_exp<<<blocks((long)E * 4, T), T, 0, stream>>>(dst, maxb, elog, den, E);
  aggregate<<<2048, T, 0, stream>>>(src, dst, C0, 512, elog, den, agg, E);
  finalize0<<<blocks((long)N * 256, T), T, 0, stream>>>(agg, C0, b0, h1, Abf1, (long)N * 256);

  // ---- layer 1 ----
  gemm_bf16_wmma<<<mblocks * (256 / 64), T, 64 * 256 * 2, stream>>>(Abf1, W1T, ft1, N, 256, 256);
  node_coeffs<<<blocks((long)N * 4, T), T, 0, stream>>>(ft1, 256, al1, ar1, el, er, N);
  fill_u32<<<blocks((long)N * 4, T), T, 0, stream>>>(maxb, 0u, (long)N * 4);
  fill_u32<<<blocks((long)N * 4, T), T, 0, stream>>>((unsigned*)den, 0u, (long)N * 4);
  fill_u32<<<blocks((long)N * 256, T), T, 0, stream>>>((unsigned*)agg, 0u, (long)N * 256);
  edge_logits<<<blocks(E, T), T, 0, stream>>>(edge_feat, Wae1, src, dst, el, er, elog, maxb, E);
  edge_exp<<<blocks((long)E * 4, T), T, 0, stream>>>(dst, maxb, elog, den, E);
  aggregate<<<2048, T, 0, stream>>>(src, dst, ft1, 256, elog, den, agg, E);
  finalize1<<<blocks((long)N * 64, T), T, 0, stream>>>(agg, h1, b1, (float*)d_out, N);
}
